// TreeFeature_83099027243625
// MI455X (gfx1250) — matrix operand order repaired and checked by reading, not compiled, not run on hardware
//
#include <hip/hip_runtime.h>
#include <hip/hip_bf16.h>
#include <math.h>

#define B_      256
#define NOBJ    100
#define NK      36
#define D_      2048
#define H2      1024
#define HID     2048
#define KS      256
#define FOUR_H2 4096

typedef __bf16 bf16;
typedef __attribute__((ext_vector_type(16))) __bf16 v16bf;
typedef __attribute__((ext_vector_type(8)))  __bf16 v8bf;
typedef __attribute__((ext_vector_type(8)))  float  v8f;

__device__ __forceinline__ float sigf(float x) { return 1.0f / (1.0f + __expf(-x)); }

// ---------------------------------------------------------------------------
// Top-K selection (JAX top_k semantics: descending, first index wins ties).
// att_top is sorted descending -> root = argmax(att_top) == slot 0 always.
// ---------------------------------------------------------------------------
__global__ void topk_kernel(const float* __restrict__ att,
                            int* __restrict__ idx_keep,
                            float* __restrict__ att_sig) {
    int b = blockIdx.x * blockDim.x + threadIdx.x;
    if (b >= B_) return;
    float v[NOBJ];
    bool  used[NOBJ];
    for (int j = 0; j < NOBJ; ++j) { v[j] = att[b * NOBJ + j]; used[j] = false; }
    for (int k = 0; k < NK; ++k) {
        float best = -INFINITY; int bi = 0;
        for (int j = 0; j < NOBJ; ++j)
            if (!used[j] && v[j] > best) { best = v[j]; bi = j; }
        used[bi] = true;
        idx_keep[b * NK + k] = bi;
        att_sig[b * NK + k]  = sigf(best);
    }
}

// Gather kept proposals into dense bf16 activation matrices [B*NK, D].
__global__ void gather_feat_kernel(const float* __restrict__ vf,
                                   const float* __restrict__ vo,
                                   const int* __restrict__ idx_keep,
                                   bf16* __restrict__ x_bf,
                                   bf16* __restrict__ vo_bf) {
    long id = (long)blockIdx.x * blockDim.x + threadIdx.x;
    const long total = (long)B_ * NK * D_;
    if (id >= total) return;
    int d = (int)(id % D_);
    int r = (int)(id / D_);        // b*NK + n
    int b = r / NK;
    int obj = idx_keep[r];
    long src = ((long)b * D_ + d) * NOBJ + obj;
    x_bf[id]  = (bf16)vf[src];
    vo_bf[id] = (bf16)vo[src];
}

__global__ void gather_box_kernel(const float* __restrict__ boxes,
                                  const int* __restrict__ idx_keep,
                                  float* __restrict__ bx_g) {
    int id = blockIdx.x * blockDim.x + threadIdx.x;
    if (id >= B_ * 4 * NK) return;
    int n = id % NK;
    int c = (id / NK) % 4;
    int b = id / (4 * NK);
    int obj = idx_keep[b * NK + n];
    bx_g[id] = boxes[(b * 4 + c) * NOBJ + obj];
}

// W[K,N] (f32) -> Wt[N,K] (bf16): GEMM consumes B transposed so each lane
// reads 16 contiguous bf16 = one B fragment slice.
__global__ void wt_convert_kernel(const float* __restrict__ W,
                                  bf16* __restrict__ Wt, int K, int N) {
    long id = (long)blockIdx.x * blockDim.x + threadIdx.x;
    if (id >= (long)K * N) return;
    int k = (int)(id % K);
    int n = (int)(id / K);
    Wt[id] = (bf16)W[(long)k * N + n];
}

// ---------------------------------------------------------------------------
// bf16 WMMA GEMM: C[M,N] = A[M,K] * Bt[N,K]^T + bias, f32 accumulate.
// Block tile 128x128, BK=32, 8 waves (2x4), wave tile 64x32 (4x2 WMMA tiles).
// LDS double-buffered; tiles staged with gfx1250 async global->LDS DMA
// (GLOBAL_LOAD_ASYNC_TO_LDS_B128, tracked by ASYNCcnt): no VGPR round-trip,
// prefetch of tile k+1 overlaps the WMMAs on tile k; one s_wait_asynccnt 0 +
// barrier per K-step closes the double buffer.
// Requires M%128==0, N%128==0, K%32==0 (true for every call site here).
// ---------------------------------------------------------------------------
__global__ __launch_bounds__(256) void gemm_bf16_kernel(
    const bf16* __restrict__ A, const bf16* __restrict__ Bt,
    float* __restrict__ C, const float* __restrict__ bias,
    int M, int N, int K) {
    __shared__ __align__(16) bf16 As[2][128][40];
    __shared__ __align__(16) bf16 Bs[2][128][40];

    const int tid   = threadIdx.x;
    const int lane  = tid & 31;
    const int wv    = tid >> 5;
    const int lm    = lane & 15;   // row(A frag) / col(B,C frag)
    const int lh    = lane >> 4;   // lane-half selector
    const int waveM = wv >> 2;     // 0..1
    const int waveN = wv & 3;      // 0..3
    const int m0 = blockIdx.y * 128;
    const int n0 = blockIdx.x * 128;

    const int lrow = tid >> 1;          // 0..127 (tile row loaded by this thread)
    const int lcol = (tid & 1) * 16;    // 0 or 16 (16 bf16 per thread)
    const int nk   = K >> 5;

    v8f acc[4][2] = {};

    const bf16* ag = A  + (long)(m0 + lrow) * K + lcol;
    const bf16* bg = Bt + (long)(n0 + lrow) * K + lcol;

    // Per-lane LDS byte offsets (low 32 bits of a generic LDS address == DS
    // offset); this thread's 2x16B slots in each of the A/B staging buffers.
    const unsigned aoff[2] = {
        (unsigned)(size_t)&As[0][lrow][lcol],
        (unsigned)(size_t)&As[1][lrow][lcol]
    };
    const unsigned boff[2] = {
        (unsigned)(size_t)&Bs[0][lrow][lcol],
        (unsigned)(size_t)&Bs[1][lrow][lcol]
    };

    // Issue the 4 async 16B global->LDS transfers for K-tile kt2 into buffer q.
    auto issue_tile = [&](int q, int kt2) {
        const bf16* ag2 = ag + kt2 * 32;
        const bf16* bg2 = bg + kt2 * 32;
        unsigned a0 = aoff[q], b0 = boff[q];
        asm volatile("global_load_async_to_lds_b128 %0, %1, off"
                     :: "v"(a0),      "v"(ag2)     : "memory");
        asm volatile("global_load_async_to_lds_b128 %0, %1, off"
                     :: "v"(a0 + 16), "v"(ag2 + 8) : "memory");
        asm volatile("global_load_async_to_lds_b128 %0, %1, off"
                     :: "v"(b0),      "v"(bg2)     : "memory");
        asm volatile("global_load_async_to_lds_b128 %0, %1, off"
                     :: "v"(b0 + 16), "v"(bg2 + 8) : "memory");
    };

    issue_tile(0, 0);
    asm volatile("s_wait_asynccnt 0x0" ::: "memory");
    __syncthreads();

    for (int kt = 0; kt < nk; ++kt) {
        const int p = kt & 1;
        if (kt + 1 < nk) issue_tile(p ^ 1, kt + 1);  // async prefetch next tile

        // B fragments: lane = column lm, K = 16*lh + 0..15 (contiguous)
        v16bf bfr[2];
#pragma unroll
        for (int j = 0; j < 2; ++j) {
            int n = waveN * 32 + j * 16 + lm;
            v8bf blo = *(const v8bf*)&Bs[p][n][lh * 16];
            v8bf bhi = *(const v8bf*)&Bs[p][n][lh * 16 + 8];
            bfr[j] = __builtin_shufflevector(blo, bhi,
                     0,1,2,3,4,5,6,7,8,9,10,11,12,13,14,15);
        }
        // A fragments: lane = row lm, K = 16*lh + 0..15 (contiguous)
#pragma unroll
        for (int i = 0; i < 4; ++i) {
            int r = waveM * 64 + i * 16 + lm;
            v8bf alo = *(const v8bf*)&As[p][r][lh * 16];
            v8bf ahi = *(const v8bf*)&As[p][r][lh * 16 + 8];
            v16bf af = __builtin_shufflevector(alo, ahi,
                       0,1,2,3,4,5,6,7,8,9,10,11,12,13,14,15);
#pragma unroll
            for (int j = 0; j < 2; ++j)
                acc[i][j] = __builtin_amdgcn_wmma_f32_16x16x32_bf16(
                    false, af, false, bfr[j], (short)0, acc[i][j], false, false);
        }

        if (kt + 1 < nk)
            asm volatile("s_wait_asynccnt 0x0" ::: "memory");
        __syncthreads();
    }

    // C/D layout: VGPR r holds M = 8*lh + r, N = lm.
#pragma unroll
    for (int j = 0; j < 2; ++j) {
        int col = n0 + waveN * 32 + j * 16 + lm;
        float bv = bias ? bias[col] : 0.0f;
#pragma unroll
        for (int i = 0; i < 4; ++i) {
            int rb = m0 + waveM * 64 + i * 16 + lh * 8;
#pragma unroll
            for (int r = 0; r < 8; ++r)
                C[(long)(rb + r) * N + col] = acc[i][j][r] + bv;
        }
    }
}

// f = tanh(voWv + bx*Wb) -- small box term + activation, in place.
__global__ void score_tanh_kernel(float* __restrict__ fpre,
                                  const float* __restrict__ bx_g,
                                  const float* __restrict__ Wb) {
    long id = (long)blockIdx.x * blockDim.x + threadIdx.x;
    if (id >= (long)B_ * NK * KS) return;
    int k = (int)(id % KS);
    int r = (int)(id / KS);
    int n = r % NK;
    int b = r / NK;
    float v = fpre[id];
#pragma unroll
    for (int c = 0; c < 4; ++c)
        v += bx_g[(b * 4 + c) * NK + n] * Wb[c * KS + k];
    fpre[id] = tanhf(v);
}

// S[b,i,j] = sigmoid(dot(f_i,f_j)/sqrt(KS)) * sig(att_i) * sig(att_j)
__global__ __launch_bounds__(256) void smat_kernel(const float* __restrict__ f,
                                                   const float* __restrict__ att_sig,
                                                   float* __restrict__ S) {
    int b = blockIdx.x;
    __shared__ float fs[NK * KS];  // 36 KB of the 320 KB WGP LDS
    for (int i = threadIdx.x; i < NK * KS; i += 256)
        fs[i] = f[(long)b * NK * KS + i];
    __syncthreads();
    for (int k = threadIdx.x; k < NK * NK; k += 256) {
        int i = k / NK, j = k % NK;
        float d = 0.0f;
        for (int q = 0; q < KS; ++q) d += fs[i * KS + q] * fs[j * KS + q];
        S[b * NK * NK + k] =
            sigf(d * (1.0f / 16.0f)) * att_sig[b * NK + i] * att_sig[b * NK + j];
    }
}

// Prim MST per batch; argmax over flat index with first-index tie-break
// (matches jnp.argmax). Root is kept-slot 0 (att_top sorted descending).
__global__ __launch_bounds__(128) void prim_kernel(const float* __restrict__ S,
                                                   int* __restrict__ parent,
                                                   int* __restrict__ order) {
    int b = blockIdx.x;
    int t = threadIdx.x;
    __shared__ float sS[NK * NK];
    __shared__ int   intree[NK];
    __shared__ float rv[128];
    __shared__ int   ri[128];
    for (int k = t; k < NK * NK; k += 128) sS[k] = S[b * NK * NK + k];
    if (t < NK) { intree[t] = (t == 0); parent[b * NK + t] = NK; }
    if (t == 0) order[b * NK] = 0;
    __syncthreads();
    for (int step = 0; step < NK - 1; ++step) {
        float bv = -INFINITY; int bi = -1;
        for (int k = t; k < NK * NK; k += 128) {
            int p = k / NK, c = k % NK;
            if (intree[p] && !intree[c]) {
                float v = sS[k];
                if (v > bv) { bv = v; bi = k; }  // strict > keeps lowest k
            }
        }
        rv[t] = bv; ri[t] = bi;
        __syncthreads();
        if (t == 0) {
            float best = -INFINITY; int bk = -1;
            for (int j = 0; j < 128; ++j) {
                if (ri[j] < 0) continue;
                if (bk < 0 || rv[j] > best || (rv[j] == best && ri[j] < bk)) {
                    best = rv[j]; bk = ri[j];
                }
            }
            int p = bk / NK, c = bk % NK;
            intree[c] = 1;
            parent[b * NK + c]        = p;
            order[b * NK + step + 1]  = c;
        }
        __syncthreads();
    }
}

// Recurrent-input gathers (one step): bf16-convert the per-batch source rows.
__global__ void gather_up_kernel(const float* __restrict__ hsum,
                                 const int* __restrict__ order, int t,
                                 bf16* __restrict__ Abuf) {
    int id = blockIdx.x * blockDim.x + threadIdx.x;
    if (id >= B_ * H2) return;
    int k = id % H2, b = id / H2;
    int node = order[b * NK + t];
    Abuf[id] = (bf16)hsum[((long)b * (NK + 1) + node) * H2 + k];
}

__global__ void gather_dn_kernel(const float* __restrict__ hd,
                                 const int* __restrict__ order,
                                 const int* __restrict__ parent, int t,
                                 bf16* __restrict__ Abuf) {
    int id = blockIdx.x * blockDim.x + threadIdx.x;
    if (id >= B_ * H2) return;
    int k = id % H2, b = id / H2;
    int node = order[b * NK + t];
    int par  = parent[b * NK + node];
    Abuf[id] = (bf16)hd[((long)b * (NK + 1) + par) * H2 + k];
}

// Child-sum LSTM cell (bottom-up) + scatter-add into parent's sums.
__global__ __launch_bounds__(256) void cell_up_kernel(
    const float* __restrict__ xWu, const float* __restrict__ rec,
    const int* __restrict__ order, const int* __restrict__ parent, int t,
    float* __restrict__ h_up, float* __restrict__ hsum, float* __restrict__ csum) {
    int b = blockIdx.x;
    int node = order[b * NK + t];
    int par  = parent[b * NK + node];
    long r36  = (long)b * NK + node;
    long prow = ((long)b * (NK + 1) + par)  * H2;
    long nrow = ((long)b * (NK + 1) + node) * H2;
    for (int j = threadIdx.x; j < H2; j += 256) {
        float i_ = xWu[r36 * FOUR_H2 + j]            + rec[(long)b * FOUR_H2 + j];
        float f_ = xWu[r36 * FOUR_H2 + H2 + j]       + rec[(long)b * FOUR_H2 + H2 + j];
        float g_ = xWu[r36 * FOUR_H2 + 2 * H2 + j]   + rec[(long)b * FOUR_H2 + 2 * H2 + j];
        float o_ = xWu[r36 * FOUR_H2 + 3 * H2 + j]   + rec[(long)b * FOUR_H2 + 3 * H2 + j];
        float cp = csum[nrow + j];
        float c  = sigf(i_) * tanhf(g_) + sigf(f_) * cp;
        float h  = sigf(o_) * tanhf(c);
        h_up[r36 * H2 + j] = h;
        hsum[prow + j] += h;   // one node per batch per step: race-free
        csum[prow + j] += c;
    }
}

// Top-down LSTM cell: child consumes parent's (h,c); sentinel row = zeros.
__global__ __launch_bounds__(256) void cell_dn_kernel(
    const float* __restrict__ xWd, const float* __restrict__ rec,
    const int* __restrict__ order, const int* __restrict__ parent, int t,
    float* __restrict__ hd, float* __restrict__ cd) {
    int b = blockIdx.x;
    int node = order[b * NK + t];
    int par  = parent[b * NK + node];
    long r36  = (long)b * NK + node;
    long prow = ((long)b * (NK + 1) + par)  * H2;
    long nrow = ((long)b * (NK + 1) + node) * H2;
    for (int j = threadIdx.x; j < H2; j += 256) {
        float i_ = xWd[r36 * FOUR_H2 + j]            + rec[(long)b * FOUR_H2 + j];
        float f_ = xWd[r36 * FOUR_H2 + H2 + j]       + rec[(long)b * FOUR_H2 + H2 + j];
        float g_ = xWd[r36 * FOUR_H2 + 2 * H2 + j]   + rec[(long)b * FOUR_H2 + 2 * H2 + j];
        float o_ = xWd[r36 * FOUR_H2 + 3 * H2 + j]   + rec[(long)b * FOUR_H2 + 3 * H2 + j];
        float cp = cd[prow + j];
        float c  = sigf(i_) * tanhf(g_) + sigf(f_) * cp;
        float h  = sigf(o_) * tanhf(c);
        hd[nrow + j] = h;
        cd[nrow + j] = c;
    }
}

// out[b, d, 0, n]: d<H2 -> h_up[b,n,d] ; else hd[b,n,d-H2]
__global__ void out_kernel(const float* __restrict__ h_up,
                           const float* __restrict__ hd,
                           float* __restrict__ out) {
    long id = (long)blockIdx.x * blockDim.x + threadIdx.x;
    const long total = (long)B_ * HID * NK;
    if (id >= total) return;
    int  n = (int)(id % NK);
    long r = id / NK;
    int  d = (int)(r % HID);
    int  b = (int)(r / HID);
    out[id] = (d < H2) ? h_up[((long)b * NK + n) * H2 + d]
                       : hd[((long)b * (NK + 1) + n) * H2 + (d - H2)];
}

// ---------------------------------------------------------------------------
extern "C" void kernel_launch(void* const* d_in, const int* in_sizes, int n_in,
                              void* d_out, int out_size, void* d_ws, size_t ws_size,
                              hipStream_t stream) {
    (void)in_sizes; (void)n_in; (void)out_size; (void)ws_size;
    const float* boxes = (const float*)d_in[0];
    const float* att   = (const float*)d_in[1];
    const float* vf    = (const float*)d_in[2];
    const float* vo    = (const float*)d_in[3];
    /* d_in[4] que_type: unused by the reference output */
    const float* Wv    = (const float*)d_in[5];
    const float* Wb    = (const float*)d_in[6];
    const float* Wxu   = (const float*)d_in[7];
    const float* Whu   = (const float*)d_in[8];
    const float* bu    = (const float*)d_in[9];
    const float* Wxd   = (const float*)d_in[10];
    const float* Whd   = (const float*)d_in[11];
    const float* bd    = (const float*)d_in[12];
    float* out = (float*)d_out;

    char* ws = (char*)d_ws;
    size_t off = 0;
    auto alloc = [&](size_t bytes) -> char* {
        char* p = ws + off;
        off += (bytes + 255) & ~(size_t)255;
        return p;
    };

    int*   idx_keep = (int*)  alloc((size_t)B_ * NK * 4);
    float* att_sig  = (float*)alloc((size_t)B_ * NK * 4);
    int*   parent   = (int*)  alloc((size_t)B_ * NK * 4);
    int*   order    = (int*)  alloc((size_t)B_ * NK * 4);
    float* bx_g     = (float*)alloc((size_t)B_ * 4 * NK * 4);
    bf16*  x_bf     = (bf16*) alloc((size_t)B_ * NK * D_ * 2);
    bf16*  vo_bf    = (bf16*) alloc((size_t)B_ * NK * D_ * 2);
    bf16*  Wv_t     = (bf16*) alloc((size_t)KS * D_ * 2);
    bf16*  Wxu_t    = (bf16*) alloc((size_t)FOUR_H2 * D_ * 2);
    bf16*  Wxd_t    = (bf16*) alloc((size_t)FOUR_H2 * D_ * 2);
    bf16*  Whu_t    = (bf16*) alloc((size_t)FOUR_H2 * H2 * 2);
    bf16*  Whd_t    = (bf16*) alloc((size_t)FOUR_H2 * H2 * 2);
    float* fpre     = (float*)alloc((size_t)B_ * NK * KS * 4);
    float* Smat     = (float*)alloc((size_t)B_ * NK * NK * 4);
    float* xWuB     = (float*)alloc((size_t)B_ * NK * FOUR_H2 * 4);
    float* xWdB     = (float*)alloc((size_t)B_ * NK * FOUR_H2 * 4);
    float* hsum     = (float*)alloc((size_t)B_ * (NK + 1) * H2 * 4);
    float* csum     = (float*)alloc((size_t)B_ * (NK + 1) * H2 * 4);
    float* h_up     = (float*)alloc((size_t)B_ * NK * H2 * 4);
    float* hd       = (float*)alloc((size_t)B_ * (NK + 1) * H2 * 4);
    float* cd       = (float*)alloc((size_t)B_ * (NK + 1) * H2 * 4);
    bf16*  Abuf     = (bf16*) alloc((size_t)B_ * H2 * 2);
    float* rec      = (float*)alloc((size_t)B_ * FOUR_H2 * 4);

    const size_t sumBytes = (size_t)B_ * (NK + 1) * H2 * 4;
    hipMemsetAsync(hsum, 0, sumBytes, stream);
    hipMemsetAsync(csum, 0, sumBytes, stream);
    hipMemsetAsync(hd,   0, sumBytes, stream);
    hipMemsetAsync(cd,   0, sumBytes, stream);

    topk_kernel<<<4, 64, 0, stream>>>(att, idx_keep, att_sig);
    {
        long tot = (long)B_ * NK * D_;
        gather_feat_kernel<<<(int)((tot + 255) / 256), 256, 0, stream>>>(
            vf, vo, idx_keep, x_bf, vo_bf);
    }
    gather_box_kernel<<<(B_ * 4 * NK + 255) / 256, 256, 0, stream>>>(boxes, idx_keep, bx_g);

    wt_convert_kernel<<<(int)(((long)D_ * KS + 255) / 256), 256, 0, stream>>>(Wv,  Wv_t,  D_, KS);
    wt_convert_kernel<<<(int)(((long)D_ * FOUR_H2 + 255) / 256), 256, 0, stream>>>(Wxu, Wxu_t, D_, FOUR_H2);
    wt_convert_kernel<<<(int)(((long)D_ * FOUR_H2 + 255) / 256), 256, 0, stream>>>(Wxd, Wxd_t, D_, FOUR_H2);
    wt_convert_kernel<<<(int)(((long)H2 * FOUR_H2 + 255) / 256), 256, 0, stream>>>(Whu, Whu_t, H2, FOUR_H2);
    wt_convert_kernel<<<(int)(((long)H2 * FOUR_H2 + 255) / 256), 256, 0, stream>>>(Whd, Whd_t, H2, FOUR_H2);

    // score features: [9216,2048] x [2048,256]
    gemm_bf16_kernel<<<dim3(KS / 128, (B_ * NK) / 128), 256, 0, stream>>>(
        vo_bf, Wv_t, fpre, nullptr, B_ * NK, KS, D_);
    score_tanh_kernel<<<(int)(((long)B_ * NK * KS + 255) / 256), 256, 0, stream>>>(fpre, bx_g, Wb);
    smat_kernel<<<B_, 256, 0, stream>>>(fpre, att_sig, Smat);
    prim_kernel<<<B_, 128, 0, stream>>>(Smat, parent, order);

    // big input projections: [9216,2048] x [2048,4096], + bias
    gemm_bf16_kernel<<<dim3(FOUR_H2 / 128, (B_ * NK) / 128), 256, 0, stream>>>(
        x_bf, Wxu_t, xWuB, bu, B_ * NK, FOUR_H2, D_);
    gemm_bf16_kernel<<<dim3(FOUR_H2 / 128, (B_ * NK) / 128), 256, 0, stream>>>(
        x_bf, Wxd_t, xWdB, bd, B_ * NK, FOUR_H2, D_);

    // bottom-up scan (reverse topological order)
    for (int tt = 0; tt < NK; ++tt) {
        int t = NK - 1 - tt;
        gather_up_kernel<<<(B_ * H2) / 256, 256, 0, stream>>>(hsum, order, t, Abuf);
        gemm_bf16_kernel<<<dim3(FOUR_H2 / 128, B_ / 128), 256, 0, stream>>>(
            Abuf, Whu_t, rec, nullptr, B_, FOUR_H2, H2);
        cell_up_kernel<<<B_, 256, 0, stream>>>(xWuB, rec, order, parent, t, h_up, hsum, csum);
    }
    // top-down scan (topological order)
    for (int t = 0; t < NK; ++t) {
        gather_dn_kernel<<<(B_ * H2) / 256, 256, 0, stream>>>(hd, order, parent, t, Abuf);
        gemm_bf16_kernel<<<dim3(FOUR_H2 / 128, B_ / 128), 256, 0, stream>>>(
            Abuf, Whd_t, rec, nullptr, B_, FOUR_H2, H2);
        cell_dn_kernel<<<B_, 256, 0, stream>>>(xWdB, rec, order, parent, t, hd, cd);
    }

    {
        long tot = (long)B_ * HID * NK;
        out_kernel<<<(int)((tot + 255) / 256), 256, 0, stream>>>(h_up, hd, out);
    }
}
